// PhoneMapper_3642132267213
// MI455X (gfx1250) — compile-verified, matrix operand order checked by reading, not run on hardware
//
#include <hip/hip_runtime.h>
#include <hip/hip_bf16.h>
#include <float.h>

// ---------------------------------------------------------------------------
// Types for CDNA5 WMMA (gfx1250, wave32)
// ---------------------------------------------------------------------------
typedef __attribute__((ext_vector_type(16))) __bf16 v16bf;
typedef __attribute__((ext_vector_type(8)))  __bf16 v8bf;
typedef __attribute__((ext_vector_type(8)))  float  v8f;

#define DDIM   1024      // feature dim (fixed by reference)
#define TOPK   10        // topn (fixed by reference)
#define CCHUNK 2048      // pool columns per workgroup chunk
#define CBLK   256       // columns per inner block (8 waves * 32)
#define MROWS  32        // query rows per workgroup (2 M-tiles)
#define NCHMAX 128       // max chunks supported by merge kernel LDS

// ---------------------------------------------------------------------------
// Prep kernels: fp32 -> bf16 conversion + reciprocal L2 norms
// ---------------------------------------------------------------------------
__global__ void prep_pool_kernel(const float* __restrict__ pool,
                                 __bf16* __restrict__ pb,
                                 float* __restrict__ pr, int N) {
  const int n   = blockIdx.x;
  const int tid = threadIdx.x;
  __shared__ float red[256];
  float s = 0.f;
  const size_t base = (size_t)n * DDIM;
  for (int j = tid; j < DDIM; j += 256) {
    float v = pool[base + j];
    pb[base + j] = (__bf16)v;
    s += v * v;
  }
  red[tid] = s;
  __syncthreads();
  for (int off = 128; off > 0; off >>= 1) {
    if (tid < off) red[tid] += red[tid + off];
    __syncthreads();
  }
  if (tid == 0) pr[n] = __frsqrt_rn(red[0]);
}

__global__ void prep_x_kernel(const float* __restrict__ x,
                              __bf16* __restrict__ xb,
                              float* __restrict__ xr, int T) {
  const int t   = blockIdx.x;   // t < Tpad
  const int tid = threadIdx.x;
  __shared__ float red[256];
  const size_t base = (size_t)t * DDIM;
  float s = 0.f;
  if (t < T) {
    for (int j = tid; j < DDIM; j += 256) {
      float v = x[base + j];
      xb[base + j] = (__bf16)v;
      s += v * v;
    }
  } else {
    for (int j = tid; j < DDIM; j += 256) xb[base + j] = (__bf16)0.f;
  }
  red[tid] = s;
  __syncthreads();
  for (int off = 128; off > 0; off >>= 1) {
    if (tid < off) red[tid] += red[tid + off];
    __syncthreads();
  }
  if (tid == 0) xr[t] = (t < T) ? __frsqrt_rn(red[0]) : 0.f;
}

// ---------------------------------------------------------------------------
// Main kernel: bf16 WMMA GEMM with 2x2 register tiling per wave
// (32 rows x 32 cols per wave -> 4 WMMAs share 8 operand loads)
// + cosine distance + per-chunk top-10.
// grid = (rowTiles, numChunks); rowTiles fastest so one 4MB pool chunk stays
// resident in L2 while all row tiles consume it.
// ---------------------------------------------------------------------------
__global__ __launch_bounds__(256)
void gemm_topk_kernel(const __bf16* __restrict__ xb,
                      const __bf16* __restrict__ pb,
                      const float* __restrict__ xr,
                      const float* __restrict__ pr,
                      float* __restrict__ cand_val,
                      int*   __restrict__ cand_idx,
                      int N, int NCH) {
  __shared__ __bf16 sA[MROWS * DDIM];        // 64 KB query tile (bf16)
  __shared__ float  sDist[MROWS][CBLK + 4];  // distance tile, padded stride
  __shared__ float  sXr[MROWS];

  const int tid     = threadIdx.x;
  const int lane    = tid & 31;
  const int wave    = tid >> 5;
  const int rowTile = blockIdx.x;
  const int chunk   = blockIdx.y;
  const int n0c     = chunk * CCHUNK;

  // --- stage query tile (32 x 1024 bf16 = 64KB) into LDS, 16B vectors ---
  {
    uint4*       dst = (uint4*)sA;
    const uint4* src = (const uint4*)(xb + (size_t)rowTile * MROWS * DDIM);
#pragma unroll
    for (int i = 0; i < 16; ++i) dst[tid + i * 256] = src[tid + i * 256];
  }
  if (tid < MROWS) sXr[tid] = xr[rowTile * MROWS + tid];
  __syncthreads();

  // WMMA fragment geometry (ISA 7.12.2, 16-bit A 16x32):
  //   lanes 0-15: M = lane,    K in {k+0..7, k+16..23}
  //   lanes 16-31:M = lane-16, K in {k+8..15, k+24..31}
  const int mL = lane & 15;
  const int kb = (lane < 16) ? 0 : 8;

  // per-row top-10 state (used by threads 0..31 only)
  float bv[TOPK];
  int   bi[TOPK];
#pragma unroll
  for (int k = 0; k < TOPK; ++k) { bv[k] = FLT_MAX; bi[k] = 0; }

  const __bf16* arow0 = sA + mL * DDIM + kb;          // M-tile 0 (rows 0..15)
  const __bf16* arow1 = sA + (16 + mL) * DDIM + kb;   // M-tile 1 (rows 16..31)

  for (int cb = 0; cb < CCHUNK / CBLK; ++cb) {
    const int n0    = n0c + cb * CBLK;      // block-first column
    const int colB  = wave * 32;            // wave's column base inside block
    const int n00   = n0 + colB + mL;       // N-tile 0 column for this lane
    const int n01   = n00 + 16;             // N-tile 1 column for this lane
    const int nc0   = (n00 < N) ? n00 : (N - 1);
    const int nc1   = (n01 < N) ? n01 : (N - 1);
    const __bf16* bbase0 = pb + (size_t)nc0 * DDIM + kb;
    const __bf16* bbase1 = pb + (size_t)nc1 * DDIM + kb;

    // prefetch next block's pool rows (global_prefetch path)
    if (cb + 1 < CCHUNK / CBLK) {
      int npf = n0 + CBLK + colB + mL;
      npf = (npf < N) ? npf : (N - 1);
      __builtin_prefetch(pb + (size_t)npf * DDIM, 0, 0);
    }

    v8f c00 = {0.f, 0.f, 0.f, 0.f, 0.f, 0.f, 0.f, 0.f};
    v8f c01 = c00, c10 = c00, c11 = c00;

#pragma unroll 8
    for (int k = 0; k < DDIM; k += 32) {
      v8bf a0lo = *(const v8bf*)(arow0 + k);
      v8bf a0hi = *(const v8bf*)(arow0 + k + 16);
      v8bf a1lo = *(const v8bf*)(arow1 + k);
      v8bf a1hi = *(const v8bf*)(arow1 + k + 16);
      v8bf b0lo = *(const v8bf*)(bbase0 + k);
      v8bf b0hi = *(const v8bf*)(bbase0 + k + 16);
      v8bf b1lo = *(const v8bf*)(bbase1 + k);
      v8bf b1hi = *(const v8bf*)(bbase1 + k + 16);
      v16bf a0, a1, b0, b1;
#pragma unroll
      for (int i = 0; i < 8; ++i) {
        a0[i] = a0lo[i]; a0[i + 8] = a0hi[i];
        a1[i] = a1lo[i]; a1[i + 8] = a1hi[i];
        b0[i] = b0lo[i]; b0[i + 8] = b0hi[i];
        b1[i] = b1lo[i]; b1[i + 8] = b1hi[i];
      }
      c00 = __builtin_amdgcn_wmma_f32_16x16x32_bf16(false, a0, false, b0,
                                                    (short)0, c00, false, false);
      c01 = __builtin_amdgcn_wmma_f32_16x16x32_bf16(false, a0, false, b1,
                                                    (short)0, c01, false, false);
      c10 = __builtin_amdgcn_wmma_f32_16x16x32_bf16(false, a1, false, b0,
                                                    (short)0, c10, false, false);
      c11 = __builtin_amdgcn_wmma_f32_16x16x32_bf16(false, a1, false, b1,
                                                    (short)0, c11, false, false);
    }

    // cosine distance + stash to LDS. C/D layout: VGPR i, lanes<16 -> M=i,
    // lanes>=16 -> M=i+8; N = lane%16.
    const float prn0 = (n00 < N) ? pr[nc0] : 0.f;
    const float prn1 = (n01 < N) ? pr[nc1] : 0.f;
    const int   rOff = (lane < 16) ? 0 : 8;
    const int   col0 = colB + mL;
    const int   col1 = col0 + 16;
#pragma unroll
    for (int i = 0; i < 8; ++i) {
      const int r0 = i + rOff;       // rows 0..15
      const int r1 = 16 + i + rOff;  // rows 16..31
      float d00 = 1.f - c00[i] * sXr[r0] * prn0;
      float d01 = 1.f - c01[i] * sXr[r0] * prn1;
      float d10 = 1.f - c10[i] * sXr[r1] * prn0;
      float d11 = 1.f - c11[i] * sXr[r1] * prn1;
      if (n00 >= N) { d00 = FLT_MAX; d10 = FLT_MAX; }
      if (n01 >= N) { d01 = FLT_MAX; d11 = FLT_MAX; }
      sDist[r0][col0] = d00;
      sDist[r0][col1] = d01;
      sDist[r1][col0] = d10;
      sDist[r1][col1] = d11;
    }
    __syncthreads();

    // 32 scan threads: per-row top-10 insertion over this 256-column block
    if (tid < MROWS) {
      for (int j = 0; j < CBLK; ++j) {
        const int n = n0 + j;
        if (n >= N) break;
        const float d = sDist[tid][j];
        if (d < bv[TOPK - 1]) {
          int p = TOPK - 1;
          while (p > 0 && bv[p - 1] > d) {
            bv[p] = bv[p - 1]; bi[p] = bi[p - 1]; --p;
          }
          bv[p] = d; bi[p] = n;
        }
      }
    }
    __syncthreads();
  }

  // write per-chunk candidates
  if (tid < MROWS) {
    const int rowG = rowTile * MROWS + tid;
    const size_t base = ((size_t)rowG * NCH + chunk) * TOPK;
#pragma unroll
    for (int k = 0; k < TOPK; ++k) {
      cand_val[base + k] = bv[k];
      cand_idx[base + k] = bi[k];
    }
  }
}

// ---------------------------------------------------------------------------
// Merge candidates -> final top-10 -> softmax(1/d) weights -> weighted mean
// ---------------------------------------------------------------------------
__global__ __launch_bounds__(256)
void merge_out_kernel(const float* __restrict__ cand_val,
                      const int*   __restrict__ cand_idx,
                      const float* __restrict__ pool,
                      float* __restrict__ out,
                      int NCH) {
  __shared__ float cv[NCHMAX * TOPK];
  __shared__ int   ci[NCHMAX * TOPK];
  __shared__ float wgt[TOPK];
  __shared__ int   sidx[TOPK];

  const int t   = blockIdx.x;
  const int tid = threadIdx.x;
  const int tot = NCH * TOPK;
  const size_t cbase = (size_t)t * NCH * TOPK;

  for (int i = tid; i < tot; i += 256) {
    cv[i] = cand_val[cbase + i];
    ci[i] = cand_idx[cbase + i];
  }
  __syncthreads();

  if (tid == 0) {
    float bv[TOPK]; int bi[TOPK];
#pragma unroll
    for (int k = 0; k < TOPK; ++k) { bv[k] = FLT_MAX; bi[k] = 0; }
    for (int i = 0; i < tot; ++i) {
      const float d = cv[i];
      if (d < bv[TOPK - 1]) {
        int p = TOPK - 1;
        while (p > 0 && bv[p - 1] > d) {
          bv[p] = bv[p - 1]; bi[p] = bi[p - 1]; --p;
        }
        bv[p] = d; bi[p] = ci[i];
      }
    }
    // softmax over 1/d  (stable), folded with the /TOPK "mean"
    float m = -FLT_MAX;
#pragma unroll
    for (int k = 0; k < TOPK; ++k) m = fmaxf(m, 1.f / bv[k]);
    float s = 0.f;
    float e[TOPK];
#pragma unroll
    for (int k = 0; k < TOPK; ++k) { e[k] = __expf(1.f / bv[k] - m); s += e[k]; }
    const float inv = 1.f / (s * (float)TOPK);
#pragma unroll
    for (int k = 0; k < TOPK; ++k) { wgt[k] = e[k] * inv; sidx[k] = bi[k]; }
  }
  __syncthreads();

  for (int d = tid; d < DDIM; d += 256) {
    float acc = 0.f;
#pragma unroll
    for (int k = 0; k < TOPK; ++k)
      acc += wgt[k] * pool[(size_t)sidx[k] * DDIM + d];
    out[(size_t)t * DDIM + d] = acc;
  }
}

// ---------------------------------------------------------------------------
// Host-side launcher
// ---------------------------------------------------------------------------
static inline size_t align256(size_t x) { return (x + 255) & ~(size_t)255; }

extern "C" void kernel_launch(void* const* d_in, const int* in_sizes, int n_in,
                              void* d_out, int out_size, void* d_ws, size_t ws_size,
                              hipStream_t stream) {
  const float* x    = (const float*)d_in[0];
  const float* pool = (const float*)d_in[1];
  (void)n_in; (void)ws_size;

  const int T    = in_sizes[0] / DDIM;                // 1500
  const int N    = in_sizes[1] / DDIM;                // 150000
  const int Tpad = ((T + MROWS - 1) / MROWS) * MROWS; // 1504
  const int rowTiles = Tpad / MROWS;                  // 47
  const int NCH  = (N + CCHUNK - 1) / CCHUNK;         // 74
  (void)out_size;

  // workspace carve-up
  char*  w   = (char*)d_ws;
  size_t off = 0;
  __bf16* pb = (__bf16*)(w + off); off = align256(off + (size_t)N * DDIM * sizeof(__bf16));
  __bf16* xb = (__bf16*)(w + off); off = align256(off + (size_t)Tpad * DDIM * sizeof(__bf16));
  float*  pr = (float*)(w + off);  off = align256(off + (size_t)N * sizeof(float));
  float*  xr = (float*)(w + off);  off = align256(off + (size_t)Tpad * sizeof(float));
  float*  cv = (float*)(w + off);  off = align256(off + (size_t)Tpad * NCH * TOPK * sizeof(float));
  int*    ci = (int*)(w + off);

  prep_pool_kernel<<<N, 256, 0, stream>>>(pool, pb, pr, N);
  prep_x_kernel<<<Tpad, 256, 0, stream>>>(x, xb, xr, T);
  gemm_topk_kernel<<<dim3(rowTiles, NCH), 256, 0, stream>>>(
      xb, pb, xr, pr, cv, ci, N, NCH);
  merge_out_kernel<<<T, 256, 0, stream>>>(cv, ci, pool, (float*)d_out, NCH);
}
